// HGNNPConv_GIB_v1_90546500534480
// MI455X (gfx1250) — matrix operand order repaired and checked by reading, not compiled, not run on hardware
//
#include <hip/hip_runtime.h>
#include <hip/hip_bf16.h>

#define NV   100000
#define ME   25000
#define NNZ  800000
#define CIN  256
#define COUT 256
#define HEADS 8

typedef __attribute__((ext_vector_type(16))) _Float16 v16h;
typedef __attribute__((ext_vector_type(8)))  _Float16 v8h;
typedef __attribute__((ext_vector_type(4)))  _Float16 v4h;
typedef __attribute__((ext_vector_type(8)))  float    v8f;

// ---------------- f32 -> f16 conversion ----------------
__global__ __launch_bounds__(256)
void conv_f16_kernel(const float* __restrict__ src, _Float16* __restrict__ dst, int n4) {
    int i = blockIdx.x * 256 + threadIdx.x;
    if (i >= n4) return;
    const float4 x = *(const float4*)(src + (long)i * 4);
    v4h o; o[0] = (_Float16)x.x; o[1] = (_Float16)x.y; o[2] = (_Float16)x.z; o[3] = (_Float16)x.w;
    *(v4h*)(dst + (long)i * 4) = o;
}

// ---------------- degree histograms ----------------
__global__ __launch_bounds__(256)
void degrees_kernel(const int* __restrict__ v_idx, const int* __restrict__ e_idx,
                    float* __restrict__ v_deg, float* __restrict__ e_deg) {
    int i = blockIdx.x * 256 + threadIdx.x;
    if (i >= NNZ) return;
    atomicAdd(v_deg + v_idx[i], 1.0f);
    atomicAdd(e_deg + e_idx[i], 1.0f);
}

// ---------------- H = X @ W^T + b via WMMA f16, 1x4 register blocking ----------------
// One wave computes a 16x64 slab of H: row tile r (6250), col group c (4 groups of 4
// tiles = 64 cols). 6250 * 4 = 25000 waves, 32 WMMAs per wave, A fragment reused 4x.
__global__ __launch_bounds__(256)
void gemm_wmma_kernel(const _Float16* __restrict__ Xh, const _Float16* __restrict__ Wh,
                      const float* __restrict__ bias, float* __restrict__ H) {
    const int wave = threadIdx.x >> 5;
    const int lane = threadIdx.x & 31;
    const int slab = blockIdx.x * 8 + wave;       // 0 .. 24999
    const int col_grp  = slab & 3;                // 4 col groups of 64
    const int row_tile = slab >> 2;               // 6250 row tiles
    if (row_tile >= NV / 16) return;              // exact grid: never taken
    const int row0 = row_tile * 16;
    const int col0 = col_grp * 64;
    const int sub  = lane & 15;                   // A row-in-tile / B col-in-tile
    const int hi   = lane >> 4;                   // 0: lanes 0-15, 1: lanes 16-31

    const _Float16* __restrict__ arow = Xh + (long)(row0 + sub) * CIN;        // X row
    const _Float16* __restrict__ b0   = Wh + (long)(col0 +  0 + sub) * CIN;   // W[co][k]
    const _Float16* __restrict__ b1   = Wh + (long)(col0 + 16 + sub) * CIN;
    const _Float16* __restrict__ b2   = Wh + (long)(col0 + 32 + sub) * CIN;
    const _Float16* __restrict__ b3   = Wh + (long)(col0 + 48 + sub) * CIN;

    v8f acc0 = {}, acc1 = {}, acc2 = {}, acc3 = {};
#pragma unroll
    for (int kb = 0; kb < CIN; kb += 32) {
        // A fragment: 16 contiguous halves per lane (shared by 4 WMMAs)
        const v8h a0 = *(const v8h*)(arow + kb + hi * 16);
        const v8h a1 = *(const v8h*)(arow + kb + hi * 16 + 8);
        const v16h a = __builtin_shufflevector(a0, a1, 0,1,2,3,4,5,6,7,8,9,10,11,12,13,14,15);
        // B fragments: 16 contiguous halves per lane (K-major within W row)
        const v16h vb0 = *(const v16h*)(b0 + kb + hi * 16);
        const v16h vb1 = *(const v16h*)(b1 + kb + hi * 16);
        const v16h vb2 = *(const v16h*)(b2 + kb + hi * 16);
        const v16h vb3 = *(const v16h*)(b3 + kb + hi * 16);
        acc0 = __builtin_amdgcn_wmma_f32_16x16x32_f16(false, a, false, vb0, (short)0, acc0, false, false);
        acc1 = __builtin_amdgcn_wmma_f32_16x16x32_f16(false, a, false, vb1, (short)0, acc1, false, false);
        acc2 = __builtin_amdgcn_wmma_f32_16x16x32_f16(false, a, false, vb2, (short)0, acc2, false, false);
        acc3 = __builtin_amdgcn_wmma_f32_16x16x32_f16(false, a, false, vb3, (short)0, acc3, false, false);
    }

    // Store 4 tiles with bias
    float* __restrict__ hrow = H + (long)(row0 + hi * 8) * COUT + col0 + sub;
    const float bb0 = bias[col0 +  0 + sub];
    const float bb1 = bias[col0 + 16 + sub];
    const float bb2 = bias[col0 + 32 + sub];
    const float bb3 = bias[col0 + 48 + sub];
#pragma unroll
    for (int r = 0; r < 8; ++r) {
        hrow[(long)r * COUT +  0] = acc0[r] + bb0;
        hrow[(long)r * COUT + 16] = acc1[r] + bb1;
        hrow[(long)r * COUT + 32] = acc2[r] + bb2;
        hrow[(long)r * COUT + 48] = acc3[r] + bb3;
    }
}

// ---------------- v->e scatter add (64 threads per nnz, 4 ch each) ----------------
__global__ __launch_bounds__(256)
void scatter_v2e_kernel(const float* __restrict__ H, const int* __restrict__ v_idx,
                        const int* __restrict__ e_idx, float* __restrict__ e_feat) {
    long t = (long)blockIdx.x * 256 + threadIdx.x;
    long nz = t >> 6;
    if (nz >= NNZ) return;
    int c = (int)(t & 63) * 4;
    int v = v_idx[nz], e = e_idx[nz];
    const float4 h = *(const float4*)(H + (long)v * COUT + c);
    float* dst = e_feat + (long)e * COUT + c;
    atomicAdd(dst + 0, h.x);
    atomicAdd(dst + 1, h.y);
    atomicAdd(dst + 2, h.z);
    atomicAdd(dst + 3, h.w);
}

// ---------------- normalize e_feat by max(|e|,1) ----------------
__global__ __launch_bounds__(256)
void norm_e_kernel(float* __restrict__ e_feat, const float* __restrict__ e_deg) {
    int i = blockIdx.x * 256 + threadIdx.x;   // M*COUT = 6.4M
    if (i >= ME * COUT) return;
    float d = e_deg[i >> 8];
    e_feat[i] *= 1.0f / fmaxf(d, 1.0f);
}

// ---------------- e->v scatter add into output accumulator ----------------
__global__ __launch_bounds__(256)
void scatter_e2v_kernel(const float* __restrict__ e_feat, const int* __restrict__ v_idx,
                        const int* __restrict__ e_idx, float* __restrict__ xacc) {
    long t = (long)blockIdx.x * 256 + threadIdx.x;
    long nz = t >> 6;
    if (nz >= NNZ) return;
    int c = (int)(t & 63) * 4;
    int v = v_idx[nz], e = e_idx[nz];
    const float4 f = *(const float4*)(e_feat + (long)e * COUT + c);
    float* dst = xacc + (long)v * COUT + c;
    atomicAdd(dst + 0, f.x);
    atomicAdd(dst + 1, f.y);
    atomicAdd(dst + 2, f.z);
    atomicAdd(dst + 3, f.w);
}

// ---------------- finalize: /v_deg, relu, attention dot, KL ----------------
__global__ __launch_bounds__(256)
void finalize_kernel(float* __restrict__ xa, const float* __restrict__ v_deg,
                     const float* __restrict__ att, float* __restrict__ kl_out) {
    const int wave = threadIdx.x >> 5;
    const int lane = threadIdx.x & 31;
    const int n = blockIdx.x * 8 + wave;
    if (n >= NV) return;
    const float inv = 1.0f / fmaxf(v_deg[n], 1.0f);
    float* __restrict__ xrow = xa + (long)n * COUT;
    const float* __restrict__ arow = att + (n & (HEADS - 1)) * COUT;
    const int c = lane * 8;
    float4 x0 = *(float4*)(xrow + c);
    float4 x1 = *(float4*)(xrow + c + 4);
    x0.x = fmaxf(x0.x * inv, 0.0f); x0.y = fmaxf(x0.y * inv, 0.0f);
    x0.z = fmaxf(x0.z * inv, 0.0f); x0.w = fmaxf(x0.w * inv, 0.0f);
    x1.x = fmaxf(x1.x * inv, 0.0f); x1.y = fmaxf(x1.y * inv, 0.0f);
    x1.z = fmaxf(x1.z * inv, 0.0f); x1.w = fmaxf(x1.w * inv, 0.0f);
    *(float4*)(xrow + c)     = x0;
    *(float4*)(xrow + c + 4) = x1;
    const float4 a0 = *(const float4*)(arow + c);
    const float4 a1 = *(const float4*)(arow + c + 4);
    float dot = x0.x * a0.x + x0.y * a0.y + x0.z * a0.z + x0.w * a0.w
              + x1.x * a1.x + x1.y * a1.y + x1.z * a1.z + x1.w * a1.w;
#pragma unroll
    for (int off = 16; off > 0; off >>= 1)
        dot += __shfl_xor(dot, off, 32);
    if (lane == 0) {
        float alpha = dot * (1.0f / (float)COUT);
        alpha = (alpha >= 0.0f) ? alpha : 0.2f * alpha;      // leaky_relu(0.2)
        float p = 1.0f / (1.0f + __expf(-alpha));            // sigmoid
        p = fminf(fmaxf(p, 0.01f), 0.99f);                   // clip
        float kl = p * __logf(2.0f * p) + (1.0f - p) * __logf(2.0f * (1.0f - p));
        atomicAdd(kl_out, kl);
    }
}

extern "C" void kernel_launch(void* const* d_in, const int* in_sizes, int n_in,
                              void* d_out, int out_size, void* d_ws, size_t ws_size,
                              hipStream_t stream) {
    const float* X     = (const float*)d_in[0];
    const int*   v_idx = (const int*)  d_in[1];
    const int*   e_idx = (const int*)  d_in[2];
    const float* W     = (const float*)d_in[3];
    const float* bias  = (const float*)d_in[4];
    const float* att   = (const float*)d_in[5];
    float* out = (float*)d_out;

    // Workspace layout (256B aligned slabs)
    char* ws = (char*)d_ws;
    _Float16* Xh     = (_Float16*)(ws + 0);              //  51,200,000 B
    _Float16* Wh     = (_Float16*)(ws + 51200000);       //     131,072 B
    float*    H      = (float*)   (ws + 51331072);       // 102,400,000 B
    float*    e_feat = (float*)   (ws + 153731072);      //  25,600,000 B
    float*    e_deg  = (float*)   (ws + 179331072);      //     100,096 B (pad)
    float*    v_deg  = (float*)   (ws + 179431168);      //     400,000 B
    // total ~179.8 MB

    // Zero accumulators + output (Xv accumulates directly into d_out; last slot = KL)
    hipMemsetAsync(e_feat, 0, (size_t)ME * COUT * sizeof(float), stream);
    hipMemsetAsync(e_deg,  0, (size_t)ME * sizeof(float), stream);
    hipMemsetAsync(v_deg,  0, (size_t)NV * sizeof(float), stream);
    hipMemsetAsync(d_out,  0, ((size_t)NV * COUT + 1) * sizeof(float), stream);

    // Convert X, W to f16
    conv_f16_kernel<<<(NV * CIN / 4 + 255) / 256, 256, 0, stream>>>(X, Xh, NV * CIN / 4);
    conv_f16_kernel<<<(COUT * CIN / 4 + 255) / 256, 256, 0, stream>>>(W, Wh, COUT * CIN / 4);

    // Degrees
    degrees_kernel<<<(NNZ + 255) / 256, 256, 0, stream>>>(v_idx, e_idx, v_deg, e_deg);

    // H = X @ W^T + b  (WMMA 1x4 blocked: 25000 waves, 8 waves/block)
    gemm_wmma_kernel<<<25000 / 8, 256, 0, stream>>>(Xh, Wh, bias, H);

    // v -> e scatter-sum, then mean
    scatter_v2e_kernel<<<(int)(((long)NNZ * 64 + 255) / 256), 256, 0, stream>>>(H, v_idx, e_idx, e_feat);
    norm_e_kernel<<<(ME * COUT + 255) / 256, 256, 0, stream>>>(e_feat, e_deg);

    // e -> v scatter-sum into d_out
    scatter_e2v_kernel<<<(int)(((long)NNZ * 64 + 255) / 256), 256, 0, stream>>>(e_feat, v_idx, e_idx, out);

    // mean + relu + attention + KL
    finalize_kernel<<<(NV + 7) / 8, 256, 0, stream>>>(out, v_deg, att, out + (size_t)NV * COUT);
}
